// GIN_46059229283235
// MI455X (gfx1250) — matrix operand order repaired and checked by reading, not compiled, not run on hardware
//
#include <hip/hip_runtime.h>

typedef __attribute__((ext_vector_type(16))) _Float16 v16h;
typedef __attribute__((ext_vector_type(8)))  float    v8f;

#define D_IN    96
#define D_NOISE 32
#define D_H     32
#define F_CAT   160
#define D_OUT   128

// ---------------------------------------------------------------------------
// Self-term init: aggX[n][0:96] = (1+ge)*X[n], aggX[n][96:128] = (1+ge)*h[n]
// ---------------------------------------------------------------------------
__global__ void init_aggX(const float* __restrict__ X, const float* __restrict__ h,
                          const float* __restrict__ ge, float* __restrict__ aggX, int N) {
    int tid = blockIdx.x * blockDim.x + threadIdx.x;
    if (tid >= N * 128) return;
    int n = tid >> 7, c = tid & 127;
    float scale = 1.0f + ge[0];
    float v = (c < D_IN) ? X[n * D_IN + c] : h[n * D_H + (c - D_IN)];
    aggX[tid] = scale * v;
}

__global__ void init_aggE(const float* __restrict__ eps, const float* __restrict__ ge,
                          float* __restrict__ aggE, int total) {
    int tid = blockIdx.x * blockDim.x + threadIdx.x;
    if (tid >= total) return;
    aggE[tid] = (1.0f + ge[0]) * eps[tid];
}

// W [160][128] f32 -> Wt [128][160] f16 (contiguous K per output channel)
__global__ void build_Wt(const float* __restrict__ W, _Float16* __restrict__ Wt, int total) {
    int tid = blockIdx.x * blockDim.x + threadIdx.x;
    if (tid >= total) return;
    int k = tid >> 7, n = tid & 127;
    Wt[n * F_CAT + k] = (_Float16)W[tid];
}

// ---------------------------------------------------------------------------
// Scatter of the shared X|h feature block: E edges x 128 feats, 4 per thread
// ---------------------------------------------------------------------------
__global__ void scatter_base(const float* __restrict__ X, const float* __restrict__ h,
                             const int* __restrict__ src, const int* __restrict__ dst,
                             float* __restrict__ aggX, int E) {
    int tid = blockIdx.x * blockDim.x + threadIdx.x;
    if (tid >= E * 32) return;
    int e = tid >> 5;
    int f = (tid & 31) << 2;                   // 0..124, 16B aligned
    int s = src[e], d = dst[e];
    const float* p = (f < D_IN) ? (X + s * D_IN + f) : (h + s * D_H + (f - D_IN));
    float4 v = *(const float4*)p;
    float* out = aggX + (size_t)d * 128 + f;
    atomicAdd(out + 0, v.x); atomicAdd(out + 1, v.y);
    atomicAdd(out + 2, v.z); atomicAdd(out + 3, v.w);
}

// Per-sample epsilon scatter: S x E edges x 32 feats, 4 per thread
__global__ void scatter_eps(const float* __restrict__ eps,
                            const int* __restrict__ src, const int* __restrict__ dst,
                            float* __restrict__ aggE, int E, int N, int total) {
    int tid = blockIdx.x * blockDim.x + threadIdx.x;
    if (tid >= total) return;
    int f = (tid & 7) << 2;
    int t = tid >> 3;
    int e = t % E;
    int ss = t / E;
    int sN = src[e], dN = dst[e];
    const float* p = eps + ((size_t)ss * N + sN) * 32 + f;
    float4 v = *(const float4*)p;
    float* out = aggE + ((size_t)ss * N + dN) * 32 + f;
    atomicAdd(out + 0, v.x); atomicAdd(out + 1, v.y);
    atomicAdd(out + 2, v.z); atomicAdd(out + 3, v.w);
}

// ---------------------------------------------------------------------------
// WMMA GEMM + fused bias/ReLU/column-mean partial reduction.
// One wave per 16x16 output tile; K = 160 = 5 x wmma_f32_16x16x32_f16.
// A-lane layout (16-bit A 16x32): element i -> K = (lane/16)*16 + i, M = lane%16
// B-lane layout: element i -> K = (lane/16)*16 + i, N = lane%16
// D layout: vgpr v -> M = v + 8*(lane/16), N = lane%16
// ---------------------------------------------------------------------------
__global__ void __launch_bounds__(128)
gemm_mean(const float* __restrict__ aggX, const float* __restrict__ aggE,
          const _Float16* __restrict__ Wt, const float* __restrict__ bias,
          float* __restrict__ means, int N, int S) {
    int wid  = blockIdx.x * (blockDim.x >> 5) + (threadIdx.x >> 5);
    int lane = threadIdx.x & 31;
    int tilesM = N >> 4;                       // 3125
    int tilesPerSample = tilesM * (D_OUT / 16);
    int s   = wid / tilesPerSample;
    int rem = wid - s * tilesPerSample;
    int mt  = rem >> 3;
    int nt  = rem & 7;

    int m  = (mt << 4) + (lane & 15);          // node row for A
    int nn = (nt << 4) + (lane & 15);          // out channel for B/D
    int hi = lane >> 4;

    const float* aggXrow = aggX + (size_t)m * 128;
    const float* aggErow = aggE + ((size_t)s * N + m) * 32;

    v8f acc = {};
    #pragma unroll
    for (int kstep = 0; kstep < 5; ++kstep) {
        int kbase = kstep * 32;
        v16h a;
        #pragma unroll
        for (int g = 0; g < 2; ++g) {
            int f0 = kbase + hi * 16 + g * 8;  // 8-run start, multiple of 8 -> never
            const float* p;                    // crosses the 96/128 region borders
            if (f0 < D_IN)       p = aggXrow + f0;
            else if (f0 < 128)   p = aggErow + (f0 - D_IN);
            else                 p = aggXrow + (f0 - 32);
            float4 lo = *(const float4*)p;
            float4 hf = *(const float4*)(p + 4);
            a[g*8+0] = (_Float16)lo.x; a[g*8+1] = (_Float16)lo.y;
            a[g*8+2] = (_Float16)lo.z; a[g*8+3] = (_Float16)lo.w;
            a[g*8+4] = (_Float16)hf.x; a[g*8+5] = (_Float16)hf.y;
            a[g*8+6] = (_Float16)hf.z; a[g*8+7] = (_Float16)hf.w;
        }
        v16h bfrag = *(const v16h*)(Wt + (size_t)nn * F_CAT + kbase + hi * 16);
        acc = __builtin_amdgcn_wmma_f32_16x16x32_f16(
                  false, a, false, bfrag, (short)0, acc, false, false);
    }

    float bv = bias[nn];
    float colsum = 0.0f;
    #pragma unroll
    for (int v = 0; v < 8; ++v)
        colsum += fmaxf(acc[v] + bv, 0.0f);    // ReLU per node, sum over this lane's 8 rows
    colsum += __shfl_xor(colsum, 16, 32);      // fold M=0..7 with M=8..15 halves
    if (lane < 16)
        atomicAdd(means + s * D_OUT + nn, colsum);
}

// Broadcast relu(mean/N) over the node axis with float4 stores
__global__ void finalize(const float* __restrict__ means, float* __restrict__ out,
                         float invN, int total4, int ND) {
    int tid = blockIdx.x * blockDim.x + threadIdx.x;
    if (tid >= total4) return;
    int idx = tid * 4;
    int s = idx / ND;
    int c = idx & 127;
    const float* mrow = means + s * D_OUT + c;
    float4 v;
    v.x = fmaxf(mrow[0] * invN, 0.0f);
    v.y = fmaxf(mrow[1] * invN, 0.0f);
    v.z = fmaxf(mrow[2] * invN, 0.0f);
    v.w = fmaxf(mrow[3] * invN, 0.0f);
    *(float4*)(out + idx) = v;
}

static inline size_t align256(size_t x) { return (x + 255) & ~(size_t)255; }

extern "C" void kernel_launch(void* const* d_in, const int* in_sizes, int n_in,
                              void* d_out, int out_size, void* d_ws, size_t ws_size,
                              hipStream_t stream) {
    const float* X   = (const float*)d_in[0];
    const float* h   = (const float*)d_in[1];
    const float* eps = (const float*)d_in[2];
    const float* W   = (const float*)d_in[3];
    const float* b   = (const float*)d_in[4];
    const float* ge  = (const float*)d_in[5];
    const int*   src = (const int*)d_in[6];
    const int*   dst = (const int*)d_in[7];

    const int N = in_sizes[0] / D_IN;                    // 50000
    const int S = in_sizes[2] / (N * D_NOISE);           // 4
    const int E = in_sizes[6];                           // 800000

    char* ws = (char*)d_ws;
    size_t offAggX  = 0;
    size_t offAggE  = align256(offAggX + (size_t)N * 128 * sizeof(float));
    size_t offWt    = align256(offAggE + (size_t)S * N * 32 * sizeof(float));
    size_t offMeans = align256(offWt   + (size_t)D_OUT * F_CAT * sizeof(_Float16));
    float*    aggX  = (float*)(ws + offAggX);
    float*    aggE  = (float*)(ws + offAggE);
    _Float16* Wt    = (_Float16*)(ws + offWt);
    float*    means = (float*)(ws + offMeans);

    hipMemsetAsync(means, 0, (size_t)S * D_OUT * sizeof(float), stream);

    {   int total = N * 128;
        init_aggX<<<(total + 255) / 256, 256, 0, stream>>>(X, h, ge, aggX, N); }
    {   int total = S * N * D_NOISE;
        init_aggE<<<(total + 255) / 256, 256, 0, stream>>>(eps, ge, aggE, total); }
    {   int total = F_CAT * D_OUT;
        build_Wt<<<(total + 255) / 256, 256, 0, stream>>>(W, Wt, total); }
    {   int total = E * 32;
        scatter_base<<<(total + 255) / 256, 256, 0, stream>>>(X, h, src, dst, aggX, E); }
    {   int total = S * E * 8;
        scatter_eps<<<(total + 255) / 256, 256, 0, stream>>>(eps, src, dst, aggE, E, N, total); }

    {   int waves = S * (N / 16) * (D_OUT / 16);         // 100000, divisible by 4
        gemm_mean<<<waves / 4, 128, 0, stream>>>(aggX, aggE, Wt, b, means, N, S); }

    {   int total4 = S * N * D_OUT / 4;
        finalize<<<(total4 + 255) / 256, 256, 0, stream>>>(means, (float*)d_out,
                                                           1.0f / (float)N, total4, N * D_OUT); }

    // epsilon passthrough (second tuple element), appended after the [S,N,128] block
    hipMemcpyAsync((float*)d_out + (size_t)S * N * D_OUT, eps,
                   (size_t)S * N * D_NOISE * sizeof(float),
                   hipMemcpyDeviceToDevice, stream);
}
